// MultiheadAttentionBlock_7481833030059
// MI455X (gfx1250) — matrix operand order repaired and checked by reading, not compiled, not run on hardware
//
#include <hip/hip_runtime.h>
#include <hip/hip_bf16.h>

// ---------------------------------------------------------------------------
// MHA block for MI455X (gfx1250), wave32 + v_wmma_f32_16x16x32_bf16.
//   out = softmax_causal((xq Wq^T + bq)(xk Wk^T + bk)^T / 8) (xv Wv^T + bv) Wo^T + bo
// ~137 GFLOP vs ~150 MB HBM -> matrix-core bound; bf16 operands, f32 accum.
// Weights pre-converted to bf16 (if workspace allows) so the GEMM hot loop
// issues pure bf16 fragment loads + WMMA with minimal VALU.
// ---------------------------------------------------------------------------

typedef __bf16 bf16_t;
typedef __attribute__((ext_vector_type(16))) __bf16 v16bf;
typedef __attribute__((ext_vector_type(8)))  __bf16 v8bf;
typedef __attribute__((ext_vector_type(8)))  float  v8f;

#define BATCH   4
#define S_LEN   2048
#define DMODEL  1024
#define NHEAD   16
#define DK      64

__device__ inline v8f zero8() {
  v8f z;
#pragma unroll
  for (int i = 0; i < 8; ++i) z[i] = 0.0f;
  return z;
}

// A fragment (16x32, MxK), bf16:
// lane L: row M = L&15 ; lanes 0-15 hold K 0-15, lanes 16-31 hold K 16-31.
__device__ inline v16bf load_a_f32(const float* base, int ld, int row0, int k0, int lane) {
  const int m  = lane & 15;
  const int kg = (lane >> 4) << 4;            // 0 or 16
  const float* p = base + (size_t)(row0 + m) * ld + k0 + kg;
  v16bf a;
#pragma unroll
  for (int i = 0; i < 16; ++i) a[i] = (__bf16)p[i];         // K = kg .. kg+15
  return a;
}

__device__ inline v16bf load_a_bf16(const bf16_t* base, int ld, int row0, int k0, int lane) {
  const int m  = lane & 15;
  const int kg = (lane >> 4) << 4;
  const bf16_t* p = base + (size_t)(row0 + m) * ld + k0 + kg;
  return *(const v16bf*)p;             // 32B aligned
}

// B fragment (32x16, KxN): source stored "transposed" (row n of source = column n
// of B, contiguous over k). lane: N = lane&15 ; K = 0-15 (lanes 0-15) / 16-31.
__device__ inline v16bf load_b_f32(const float* base, int ld, int n0, int k0, int lane) {
  const int n    = lane & 15;
  const int koff = (lane >> 4) << 4;          // 0 or 16
  const float* p = base + (size_t)(n0 + n) * ld + k0 + koff;
  v16bf b;
#pragma unroll
  for (int i = 0; i < 16; ++i) b[i] = (__bf16)p[i];
  return b;
}

__device__ inline v16bf load_b_bf16(const bf16_t* base, int ld, int n0, int k0, int lane) {
  const int n    = lane & 15;
  const int koff = (lane >> 4) << 4;
  const bf16_t* p = base + (size_t)(n0 + n) * ld + k0 + koff;
  return *(const v16bf*)p;            // 32B aligned
}

__device__ inline v8f wmma_bf16(v16bf a, v16bf b, v8f c) {
  return __builtin_amdgcn_wmma_f32_16x16x32_bf16(false, a, false, b, (short)0, c,
                                                 false, false);
}

// ---------------------------------------------------------------------------
// f32 -> bf16 weight conversion (one-shot, 1M elements per weight matrix).
// ---------------------------------------------------------------------------
__global__ __launch_bounds__(256) void cvt_f32_to_bf16(const float* __restrict__ src,
                                                       bf16_t* __restrict__ dst) {
  const int i = (blockIdx.x * 256 + threadIdx.x) * 4;
  const float4 f = *(const float4*)(src + i);
  v8bf* d4 = nullptr; (void)d4;
  bf16_t out[4] = {(bf16_t)f.x, (bf16_t)f.y, (bf16_t)f.z, (bf16_t)f.w};
  *(unsigned long long*)(dst + i) = *(const unsigned long long*)out;
}

// ---------------------------------------------------------------------------
// GEMM: C[M=8192, N=1024] = A[M,1024] @ W[N,1024]^T + bias[N]
// block = 256 threads (8 waves, 4(M) x 2(N)); block tile 128x128; wave tile 32x64.
// MODE 0: store bf16 to [B,H,S,DK]   (Q, K)
// MODE 1: store bf16 to [B,H,DK,S]   (V transposed -> contiguous attention B-frags)
// MODE 2: store f32  to [M,N] flat   (final output)
// ---------------------------------------------------------------------------
template <int MODE, bool A_BF16, bool W_BF16>
__global__ __launch_bounds__(256) void gemm_xwT(const void* __restrict__ Av,
                                                const void* __restrict__ Wv,
                                                const float* __restrict__ bias,
                                                void* __restrict__ Out) {
  const int lane = threadIdx.x & 31;
  const int wave = threadIdx.x >> 5;
  const int wm = wave & 3;
  const int wn = wave >> 2;
  const int row0 = blockIdx.x * 128 + wm * 32;
  const int col0 = blockIdx.y * 128 + wn * 64;

  v8f acc[2][4];
#pragma unroll
  for (int i = 0; i < 2; ++i)
#pragma unroll
    for (int j = 0; j < 4; ++j) acc[i][j] = zero8();

  for (int k0 = 0; k0 < DMODEL; k0 += 32) {
    // Prefetch next k-tile (lands in cache while this tile's WMMAs run).
    if (k0 + 32 < DMODEL) {
      const int m = lane & 15;
      if (A_BF16) {
        __builtin_prefetch((const bf16_t*)Av + (size_t)(row0 + m) * DMODEL + k0 + 32, 0, 3);
      } else {
        __builtin_prefetch((const float*)Av + (size_t)(row0 + m) * DMODEL + k0 + 32, 0, 3);
      }
      if (W_BF16) {
        __builtin_prefetch((const bf16_t*)Wv + (size_t)(col0 + m) * DMODEL + k0 + 32, 0, 3);
      } else {
        __builtin_prefetch((const float*)Wv + (size_t)(col0 + m) * DMODEL + k0 + 32, 0, 3);
      }
    }

    v16bf a0, a1;
    if (A_BF16) {
      const bf16_t* A = (const bf16_t*)Av;
      a0 = load_a_bf16(A, DMODEL, row0, k0, lane);
      a1 = load_a_bf16(A, DMODEL, row0 + 16, k0, lane);
    } else {
      const float* A = (const float*)Av;
      a0 = load_a_f32(A, DMODEL, row0, k0, lane);
      a1 = load_a_f32(A, DMODEL, row0 + 16, k0, lane);
    }
#pragma unroll
    for (int t = 0; t < 4; ++t) {
      v16bf bfr;
      if (W_BF16) bfr = load_b_bf16((const bf16_t*)Wv, DMODEL, col0 + t * 16, k0, lane);
      else        bfr = load_b_f32((const float*)Wv, DMODEL, col0 + t * 16, k0, lane);
      acc[0][t] = wmma_bf16(a0, bfr, acc[0][t]);
      acc[1][t] = wmma_bf16(a1, bfr, acc[1][t]);
    }
  }

  // C layout: lane n<16 -> N=n, M=r ; lane>=16 -> N=lane-16, M=8+r.
  const int ncol  = lane & 15;
  const int mbase = (lane >> 4) << 3;
#pragma unroll
  for (int mt = 0; mt < 2; ++mt) {
#pragma unroll
    for (int t = 0; t < 4; ++t) {
#pragma unroll
      for (int r = 0; r < 8; ++r) {
        const int m = row0 + mt * 16 + mbase + r;
        const int n = col0 + t * 16 + ncol;
        const float val = acc[mt][t][r] + bias[n];
        if (MODE == 2) {
          ((float*)Out)[(size_t)m * DMODEL + n] = val;
        } else {
          const int b = m >> 11;            // m / S_LEN
          const int s = m & (S_LEN - 1);
          const int h = n >> 6;             // n / DK
          const int d = n & (DK - 1);
          if (MODE == 0) {
            ((bf16_t*)Out)[(((size_t)(b * NHEAD + h)) * S_LEN + s) * DK + d] = (bf16_t)val;
          } else {
            ((bf16_t*)Out)[(((size_t)(b * NHEAD + h)) * DK + d) * S_LEN + s] = (bf16_t)val;
          }
        }
      }
    }
  }
}

// ---------------------------------------------------------------------------
// Flash attention, causal. grid = (S/64, B*H), block = 128 (4 waves).
// Each wave owns 16 query rows; key blocks of 32; online softmax; P transposed
// C-layout -> A-layout through a private LDS tile per wave.
// Loop counts are block-uniform so __syncthreads() is legal; fully-masked
// tiles contribute exp(-1e30 - m) == 0.
// ---------------------------------------------------------------------------
__global__ __launch_bounds__(128) void attn_kernel(const bf16_t* __restrict__ Qb,
                                                   const bf16_t* __restrict__ Kb,
                                                   const bf16_t* __restrict__ Vt,
                                                   bf16_t* __restrict__ Ctx) {
  const int lane = threadIdx.x & 31;
  const int wave = threadIdx.x >> 5;
  const int bh = blockIdx.y;
  const int b = bh >> 4;
  const int h = bh & 15;
  const bf16_t* Q = Qb + (size_t)bh * S_LEN * DK;
  const bf16_t* K = Kb + (size_t)bh * S_LEN * DK;
  const bf16_t* V = Vt + (size_t)bh * DK * S_LEN;   // [DK, S]

  const int q0    = blockIdx.x * 64 + wave * 16;
  const int ncol  = lane & 15;
  const int mrow  = (lane >> 4) << 3;

  __shared__ bf16_t lds[4][16][40];   // 16x32 P tile per wave, padded rows

  const v16bf qf0 = load_a_bf16(Q, DK, q0, 0, lane);
  const v16bf qf1 = load_a_bf16(Q, DK, q0, 32, lane);

  float mstat[8], lstat[8];
  v8f acc[4];
#pragma unroll
  for (int r = 0; r < 8; ++r) { mstat[r] = -1e30f; lstat[r] = 0.0f; }
#pragma unroll
  for (int nd = 0; nd < 4; ++nd) acc[nd] = zero8();

  const int kend = blockIdx.x * 64 + 64;            // uniform over the block
  for (int j0 = 0; j0 < kend; j0 += 32) {
    // prefetch next K block while this block's WMMAs/softmax run
    if (j0 + 32 < kend) {
      __builtin_prefetch(K + (size_t)(j0 + 32 + (lane & 15)) * DK, 0, 3);
    }
    const v8f z = zero8();
    // scores S = Q (16x64) . K^T (64x32), two 16x16 tiles, K accumulated over d.
    v8f s0 = wmma_bf16(qf1, load_b_bf16(K, DK, j0,      32, lane),
             wmma_bf16(qf0, load_b_bf16(K, DK, j0,       0, lane), z));
    v8f s1 = wmma_bf16(qf1, load_b_bf16(K, DK, j0 + 16, 32, lane),
             wmma_bf16(qf0, load_b_bf16(K, DK, j0 + 16,  0, lane), z));

    float p0[8], p1[8];
#pragma unroll
    for (int r = 0; r < 8; ++r) {
      const int qi = q0 + mrow + r;
      float v0 = (j0 + ncol      <= qi) ? s0[r] * 0.125f : -1e30f;
      float v1 = (j0 + 16 + ncol <= qi) ? s1[r] * 0.125f : -1e30f;
      float rmax = fmaxf(v0, v1);
#pragma unroll
      for (int d = 1; d < 16; d <<= 1) rmax = fmaxf(rmax, __shfl_xor(rmax, d, 32));
      const float mn   = fmaxf(mstat[r], rmax);
      const float corr = __expf(mstat[r] - mn);
      const float e0   = __expf(v0 - mn);
      const float e1   = __expf(v1 - mn);
      float rsum = e0 + e1;
#pragma unroll
      for (int d = 1; d < 16; d <<= 1) rsum += __shfl_xor(rsum, d, 32);
      lstat[r] = lstat[r] * corr + rsum;
      mstat[r] = mn;
#pragma unroll
      for (int nd = 0; nd < 4; ++nd) acc[nd][r] *= corr;
      p0[r] = e0; p1[r] = e1;
    }

    // P: C-layout -> LDS -> A-layout
    __syncthreads();
#pragma unroll
    for (int r = 0; r < 8; ++r) {
      lds[wave][mrow + r][ncol]      = (bf16_t)p0[r];
      lds[wave][mrow + r][16 + ncol] = (bf16_t)p1[r];
    }
    __syncthreads();

    v16bf pf;
    {
      const int m  = lane & 15;
      const int kg = (lane >> 4) << 4;
#pragma unroll
      for (int i = 0; i < 16; ++i) {
        pf[i]     = lds[wave][m][kg + i];
      }
    }

    // ctx += P (16x32) . V (32x64); V^T layout makes B-frags contiguous per lane.
#pragma unroll
    for (int nd = 0; nd < 4; ++nd) {
      v16bf vf = load_b_bf16(V, S_LEN, nd * 16, j0, lane);
      acc[nd] = wmma_bf16(pf, vf, acc[nd]);
    }
  }

  // normalize and store merged-head ctx as bf16 [B, S, DMODEL]
#pragma unroll
  for (int r = 0; r < 8; ++r) {
    const float inv = 1.0f / lstat[r];
    const int s = q0 + mrow + r;
#pragma unroll
    for (int nd = 0; nd < 4; ++nd) {
      const float val = acc[nd][r] * inv;
      Ctx[((size_t)b * S_LEN + s) * DMODEL + h * DK + nd * 16 + ncol] = (bf16_t)val;
    }
  }
}

// ---------------------------------------------------------------------------
extern "C" void kernel_launch(void* const* d_in, const int* in_sizes, int n_in,
                              void* d_out, int out_size, void* d_ws, size_t ws_size,
                              hipStream_t stream) {
  const float* q  = (const float*)d_in[0];
  const float* k  = (const float*)d_in[1];
  const float* v  = (const float*)d_in[2];
  const float* wq = (const float*)d_in[3];
  const float* bq = (const float*)d_in[4];
  const float* wk = (const float*)d_in[5];
  const float* bk = (const float*)d_in[6];
  const float* wv = (const float*)d_in[7];
  const float* bv = (const float*)d_in[8];
  const float* wo = (const float*)d_in[9];
  const float* bo = (const float*)d_in[10];
  // d_in[11] = causal mask; implemented analytically (exact tril).

  const size_t elems = (size_t)BATCH * S_LEN * DMODEL;     // 8M per buffer
  const size_t welems = (size_t)DMODEL * DMODEL;           // 1M per weight
  bf16_t* Qb  = (bf16_t*)d_ws;
  bf16_t* Kb  = Qb + elems;
  bf16_t* Vt  = Kb + elems;
  bf16_t* Ctx = Vt + elems;                                // base: 64 MB

  const size_t base_bytes = 4 * elems * sizeof(bf16_t);
  const bool wbf = ws_size >= base_bytes + 4 * welems * sizeof(bf16_t); // +8 MB

  const dim3 gg(64, 8);
  const dim3 gb(256);

  if (wbf) {
    bf16_t* Wqb = Ctx + elems;
    bf16_t* Wkb = Wqb + welems;
    bf16_t* Wvb = Wkb + welems;
    bf16_t* Wob = Wvb + welems;
    const int cvtBlocks = (int)(welems / (256 * 4));       // 1024
    cvt_f32_to_bf16<<<cvtBlocks, 256, 0, stream>>>(wq, Wqb);
    cvt_f32_to_bf16<<<cvtBlocks, 256, 0, stream>>>(wk, Wkb);
    cvt_f32_to_bf16<<<cvtBlocks, 256, 0, stream>>>(wv, Wvb);
    cvt_f32_to_bf16<<<cvtBlocks, 256, 0, stream>>>(wo, Wob);

    gemm_xwT<0, false, true><<<gg, gb, 0, stream>>>((const void*)q, (const void*)Wqb, bq, (void*)Qb);
    gemm_xwT<0, false, true><<<gg, gb, 0, stream>>>((const void*)k, (const void*)Wkb, bk, (void*)Kb);
    gemm_xwT<1, false, true><<<gg, gb, 0, stream>>>((const void*)v, (const void*)Wvb, bv, (void*)Vt);
    attn_kernel<<<dim3(S_LEN / 64, BATCH * NHEAD), 128, 0, stream>>>(Qb, Kb, Vt, Ctx);
    gemm_xwT<2, true, true><<<gg, gb, 0, stream>>>((const void*)Ctx, (const void*)Wob, bo, d_out);
  } else {
    gemm_xwT<0, false, false><<<gg, gb, 0, stream>>>((const void*)q, (const void*)wq, bq, (void*)Qb);
    gemm_xwT<0, false, false><<<gg, gb, 0, stream>>>((const void*)k, (const void*)wk, bk, (void*)Kb);
    gemm_xwT<1, false, false><<<gg, gb, 0, stream>>>((const void*)v, (const void*)wv, bv, (void*)Vt);
    attn_kernel<<<dim3(S_LEN / 64, BATCH * NHEAD), 128, 0, stream>>>(Qb, Kb, Vt, Ctx);
    gemm_xwT<2, true, false><<<gg, gb, 0, stream>>>((const void*)Ctx, (const void*)wo, bo, d_out);
  }
}
